// _InteractionBlock_30107720745193
// MI455X (gfx1250) — compile-verified, matrix operand order checked by reading, not run on hardware
//
#include <hip/hip_runtime.h>

typedef __attribute__((ext_vector_type(16))) _Float16 v16h;
typedef __attribute__((ext_vector_type(8)))  _Float16 v8h;
typedef __attribute__((ext_vector_type(8)))  float    v8f;
typedef __attribute__((ext_vector_type(4)))  float    v4f;

#define MNBR 32
#define HIDD 128
#define NBRF 64

// packed-weight layout (units: halfs). frag(kt,j,lane) = 16 contiguous halfs (32B).
#define FW1P_OFF 0                       // 2 ktiles * 8 jtiles * 32 lanes * 16 = 8192
#define FW2P_OFF 8192                    // 4*8*32*16 = 16384
#define UW1P_OFF (8192 + 16384)
#define UW2P_OFF (8192 + 2 * 16384)
#define WPACK_HALFS (8192 + 3 * 16384)   // 57344 halfs = 112 KB

// LDS layout (bytes)
#define WBUF_BYTES  (WPACK_HALFS * 2)        // 114688
#define HBUF_OFF    WBUF_BYTES               // 16 waves * 16*128*2B = 65536
#define MSG_OFF     (WBUF_BYTES + 65536)     // 16*128*4B = 8192
#define LDS_BYTES   (MSG_OFF + 8192)         // 188416

__device__ __forceinline__ float silu_f(float x) {
    return x / (1.0f + __expf(-x));
}

__device__ __forceinline__ v8f wmma_f16(v16h a, v16h b, v8f c) {
    return __builtin_amdgcn_wmma_f32_16x16x32_f16(false, a, false, b,
                                                  (short)0, c, false, false);
}

// Repack W[K x 128] (row-major fp32) -> f16 B-fragments.
// Fragment layout (16-bit 32x16 B, wave32): lane holds col n = 16*j + (lane&15),
// halfs[0..7] = K = 32*kt + c0 + i, halfs[8..15] = K = 32*kt + c0 + 16 + i,
// with c0 = (lane<16 ? 0 : 8).
__global__ void repack_weight_kernel(const float* __restrict__ src,
                                     _Float16* __restrict__ dst, int ktiles) {
    int t = blockIdx.x * blockDim.x + threadIdx.x;
    int total = ktiles * 8 * 32;
    if (t >= total) return;
    int lane = t & 31;
    int j    = (t >> 5) & 7;
    int kt   = t >> 8;
    int n  = 16 * j + (lane & 15);
    int c0 = 32 * kt + ((lane & 16) ? 8 : 0);
    _Float16* o = dst + ((size_t)((kt << 3) + j) * 32 + lane) * 16;
#pragma unroll
    for (int i = 0; i < 8; ++i) {
        o[i]     = (_Float16)src[(c0 + i) * HIDD + n];
        o[8 + i] = (_Float16)src[(c0 + 16 + i) * HIDD + n];
    }
}

// Fused interaction block. Block = 16 nodes (512 (node,m) rows), 512 threads.
// Wave w owns node (blk*16 + w): strip 0 = neighbors 0..15, strip 1 = 16..31.
__global__ void __launch_bounds__(512) interaction_kernel(
    const float* __restrict__ node_repr,       // [N,128]
    const float* __restrict__ nbr_fea,         // [N,32,64]
    const long long* __restrict__ nbr_idx,     // [N,32]
    const float* __restrict__ fb1,
    const float* __restrict__ fb2,
    const float* __restrict__ ub1,
    const float* __restrict__ ub2,
    const _Float16* __restrict__ wpack,
    float* __restrict__ out)                   // [N,128]
{
    extern __shared__ char smem[];
    _Float16* s_w   = (_Float16*)smem;
    float*    s_msg = (float*)(smem + MSG_OFF);

    const int tid  = threadIdx.x;
    const int wave = tid >> 5;
    const int lane = tid & 31;
    const int l16  = lane & 15;
    const int hi   = lane >> 4;        // 0 or 1
    const int blk  = blockIdx.x;

    // --- stage packed weights into LDS (112 KB), zero message buffer ---
    {
        const uint4* src = (const uint4*)wpack;
        uint4* dst = (uint4*)s_w;
        const int n16 = WBUF_BYTES / 16;
        for (int i = tid; i < n16; i += 512) dst[i] = src[i];
    }
    for (int i = tid; i < 16 * HIDD; i += 512) s_msg[i] = 0.0f;
    __syncthreads();

    // per-lane bias values for each j-tile (col = 16*j + l16)
    float b1j[8], b2j[8];
#pragma unroll
    for (int j = 0; j < 8; ++j) {
        b1j[j] = fb1[16 * j + l16];
        b2j[j] = fb2[16 * j + l16];
    }

    _Float16* s_h = (_Float16*)(smem + HBUF_OFF + wave * 4096); // 16x128 f16, per-wave

    auto bfrag = [&](int baseOff, int kt, int j) -> v16h {
        return *(const v16h*)(s_w + baseOff + (((kt << 3) + j) * 32 + lane) * 16);
    };

    const int c0h = hi ? 8 : 0;
    const long long nodeg = (long long)blk * 16 + wave;

    float pmsum[8];
#pragma unroll
    for (int j = 0; j < 8; ++j) pmsum[j] = 0.0f;

    // --- main phase: 2 strips per wave; strip s = neighbors [16s, 16s+16) of node ---
#pragma unroll
    for (int s = 0; s < 2; ++s) {
        const int strip = wave * 2 + s;         // 0..31
        const long long flat_row = (long long)blk * 512 + strip * 16 + l16;
        const float* arow = nbr_fea + flat_row * NBRF;

        // A1 fragments: 16x64 f16, 2 k-tiles
        v16h a1[2];
#pragma unroll
        for (int kt = 0; kt < 2; ++kt) {
            const int c0 = 32 * kt + c0h;
            v4f x0 = *(const v4f*)(arow + c0);
            v4f x1 = *(const v4f*)(arow + c0 + 4);
            v4f y0 = *(const v4f*)(arow + c0 + 16);
            v4f y1 = *(const v4f*)(arow + c0 + 20);
            v16h a;
#pragma unroll
            for (int i = 0; i < 4; ++i) {
                a[i]      = (_Float16)x0[i];
                a[4 + i]  = (_Float16)x1[i];
                a[8 + i]  = (_Float16)y0[i];
                a[12 + i] = (_Float16)y1[i];
            }
            a1[kt] = a;
        }

        // GEMM1: h = silu(A1 @ fw1 + fb1). kt-outer/j-inner: 8 independent accs,
        // no WMMA->WMMA RAW chains, LDS fragment loads pipeline ahead.
        v8f acc1[8] = {};
#pragma unroll
        for (int kt = 0; kt < 2; ++kt)
#pragma unroll
            for (int j = 0; j < 8; ++j)
                acc1[j] = wmma_f16(a1[kt], bfrag(FW1P_OFF, kt, j), acc1[j]);

#pragma unroll
        for (int j = 0; j < 8; ++j) {
            const int col = 16 * j + l16;
#pragma unroll
            for (int v = 0; v < 8; ++v) {
                const int row = v + 8 * hi;
                s_h[row * HIDD + col] = (_Float16)silu_f(acc1[j][v] + b1j[j]);
            }
        }
        // wave-local transpose: DS ops are in-order within a wave; this is just a
        // compiler scheduling fence (no ISA cost, no workgroup serialization).
        __builtin_amdgcn_wave_barrier();

        // A2 fragments from h: 16x128 f16, 4 k-tiles (row = l16)
        v16h a2[4];
        const _Float16* hrow = s_h + l16 * HIDD;
#pragma unroll
        for (int kt = 0; kt < 4; ++kt) {
            const int c0 = 32 * kt + c0h;
            v8h lo  = *(const v8h*)(hrow + c0);
            v8h hi8 = *(const v8h*)(hrow + c0 + 16);
            v16h a;
#pragma unroll
            for (int i = 0; i < 8; ++i) { a[i] = lo[i]; a[8 + i] = hi8[i]; }
            a2[kt] = a;
        }

        // GEMM2: filters = A2 @ fw2 + fb2, independent accumulators
        v8f acc2[8] = {};
#pragma unroll
        for (int kt = 0; kt < 4; ++kt)
#pragma unroll
            for (int j = 0; j < 8; ++j)
                acc2[j] = wmma_f16(a2[kt], bfrag(FW2P_OFF, kt, j), acc2[j]);

        // fused gather * filters, reduce over this half-group's 8 rows
        // (rows are m = 16*s + v + 8*hi of this wave's node)
        const long long* ip = nbr_idx + nodeg * MNBR + (s << 4) + 8 * hi;
        const float* gptr[8];
#pragma unroll
        for (int v = 0; v < 8; ++v) gptr[v] = node_repr + ip[v] * HIDD;

#pragma unroll
        for (int j = 0; j < 8; ++j) {
            const int col = 16 * j + l16;
            float pm = 0.0f;
#pragma unroll
            for (int v = 0; v < 8; ++v)
                pm += (acc2[j][v] + b2j[j]) * gptr[v][col];
            pmsum[j] += pm;
        }
        // protect per-wave s_h reuse in next strip (compiler fence only)
        __builtin_amdgcn_wave_barrier();
    }

    // one atomic per (j): only the two hi-groups of this wave contend per address
#pragma unroll
    for (int j = 0; j < 8; ++j)
        atomicAdd(&s_msg[wave * HIDD + 16 * j + l16], pmsum[j]);

    __syncthreads();   // all messages complete before update MLP

    // --- update MLP on the 16 node messages: 8 waves, one j-tile each ---
    _Float16* s_u = (_Float16*)(smem + HBUF_OFF);   // reuse 4KB of h staging

    if (wave < 8) {
        const int j = wave;
        const float* mrow = s_msg + l16 * HIDD;
        v16h a3[4];
#pragma unroll
        for (int kt = 0; kt < 4; ++kt) {
            const int c0 = 32 * kt + c0h;
            v16h a;
#pragma unroll
            for (int i = 0; i < 8; ++i) {
                a[i]     = (_Float16)mrow[c0 + i];
                a[8 + i] = (_Float16)mrow[c0 + 16 + i];
            }
            a3[kt] = a;
        }
        v8f acc = {};
#pragma unroll
        for (int kt = 0; kt < 4; ++kt)
            acc = wmma_f16(a3[kt], bfrag(UW1P_OFF, kt, j), acc);
        const int col = 16 * j + l16;
        const float bu1 = ub1[col];
#pragma unroll
        for (int v = 0; v < 8; ++v) {
            const int row = v + 8 * hi;
            s_u[row * HIDD + col] = (_Float16)silu_f(acc[v] + bu1);
        }
    }
    __syncthreads();

    if (wave < 8) {
        const int j = wave;
        const _Float16* urow = s_u + l16 * HIDD;
        v16h a4[4];
#pragma unroll
        for (int kt = 0; kt < 4; ++kt) {
            const int c0 = 32 * kt + c0h;
            v8h lo  = *(const v8h*)(urow + c0);
            v8h hi8 = *(const v8h*)(urow + c0 + 16);
            v16h a;
#pragma unroll
            for (int i = 0; i < 8; ++i) { a[i] = lo[i]; a[8 + i] = hi8[i]; }
            a4[kt] = a;
        }
        v8f acc = {};
#pragma unroll
        for (int kt = 0; kt < 4; ++kt)
            acc = wmma_f16(a4[kt], bfrag(UW2P_OFF, kt, j), acc);
        const int col = 16 * j + l16;
        const float bu2 = ub2[col];
#pragma unroll
        for (int v = 0; v < 8; ++v) {
            const long long row_g = (long long)blk * 16 + v + 8 * hi;
            out[row_g * HIDD + col] = acc[v] + bu2;
        }
    }
}

extern "C" void kernel_launch(void* const* d_in, const int* in_sizes, int n_in,
                              void* d_out, int out_size, void* d_ws, size_t ws_size,
                              hipStream_t stream) {
    const float*     node_repr = (const float*)d_in[0];
    const float*     nbr_fea   = (const float*)d_in[1];
    const long long* nbr_idx   = (const long long*)d_in[2];
    const float*     fw1 = (const float*)d_in[3];
    const float*     fb1 = (const float*)d_in[4];
    const float*     fw2 = (const float*)d_in[5];
    const float*     fb2 = (const float*)d_in[6];
    const float*     uw1 = (const float*)d_in[7];
    const float*     ub1 = (const float*)d_in[8];
    const float*     uw2 = (const float*)d_in[9];
    const float*     ub2 = (const float*)d_in[10];
    float* out = (float*)d_out;

    _Float16* wpack = (_Float16*)d_ws;

    const int nnode = in_sizes[0] / HIDD;       // 50000
    const int nblk  = nnode / 16;               // 3125

    // pack weights to f16 fragment order (deterministic, re-run every call)
    repack_weight_kernel<<<dim3((2 * 8 * 32 + 255) / 256), 256, 0, stream>>>(
        fw1, wpack + FW1P_OFF, 2);
    repack_weight_kernel<<<dim3((4 * 8 * 32 + 255) / 256), 256, 0, stream>>>(
        fw2, wpack + FW2P_OFF, 4);
    repack_weight_kernel<<<dim3((4 * 8 * 32 + 255) / 256), 256, 0, stream>>>(
        uw1, wpack + UW1P_OFF, 4);
    repack_weight_kernel<<<dim3((4 * 8 * 32 + 255) / 256), 256, 0, stream>>>(
        uw2, wpack + UW2P_OFF, 4);

    interaction_kernel<<<dim3(nblk), 512, LDS_BYTES, stream>>>(
        node_repr, nbr_fea, nbr_idx, fb1, fb2, ub1, ub2, wpack, out);
}
